// Connectivity3D_62294205661200
// MI455X (gfx1250) — matrix operand order repaired and checked by reading, not compiled, not run on hardware
//
#include <hip/hip_runtime.h>

typedef __attribute__((ext_vector_type(16))) __bf16 v16bf;
typedef __attribute__((ext_vector_type(8)))  __bf16 v8bf;
typedef __attribute__((ext_vector_type(8)))  float  v8f;

#define PPTS 512

static __device__ __forceinline__ v8f wmma_bf(v16bf a, v16bf b, v8f c){
  // D = A(16x32 bf16) * B(32x16 bf16) + C(16x16 f32)
  return __builtin_amdgcn_wmma_f32_16x16x32_bf16(false, a, false, b, (short)0, c, false, false);
}

// ---------------- weight repack: f32 row-major -> bf16 B-fragment order ----------------
// dst[kc][nt][lane][e], lane = half*16+lr : col = nt*16+lr, K = kc*32 + half*8 + (e&7) + (e>>3)*16
__global__ void pack_b_kernel(__bf16* __restrict__ dst, const float* __restrict__ W,
                              const float* __restrict__ Wadd,
                              int Krows, int ntiles, int Ncol, int total)
{
  int idx = blockIdx.x * 256 + threadIdx.x;
  if (idx >= total) return;
  int e    = idx & 15;
  int lane = (idx >> 4) & 31;
  int nt   = (idx >> 9) % ntiles;
  int kc   = idx / (512 * ntiles);
  int K    = kc * 32 + (lane >> 4) * 8 + (e & 7) + (e >> 3) * 16;
  int col  = nt * 16 + (lane & 15);
  float v = 0.f;
  if (K < Krows) {
    v = W[K * Ncol + col];
    if (Wadd) v += Wadd[K * Ncol + col];
  }
  dst[idx] = (__bf16)v;
}

// ---------------- PointNet: per-node 512x(6->64->128->256) + max-pool ----------------
__global__ __launch_bounds__(256) void pointnet_kernel(
  const float* __restrict__ pcls,
  const float* __restrict__ b1, const float* __restrict__ g1, const float* __restrict__ bt1,
  const float* __restrict__ b2, const float* __restrict__ g2, const float* __restrict__ bt2,
  const float* __restrict__ b3, const float* __restrict__ g3, const float* __restrict__ bt3,
  const uint4* __restrict__ wB,
  float* __restrict__ feat)
{
  extern __shared__ char smem[];
  __bf16* B1s   = (__bf16*)smem;          // [4][32][16]          2048
  __bf16* B2s   = B1s + 2048;             // [2][8][32][16]       8192
  __bf16* B3s   = B2s + 8192;             // [4][16][32][16]      32768
  __bf16* hball = B3s + 32768;            // 8 waves * [16][128]  16384
  float*  featb = (float*)(hball + 8*2048); // 8*256 f32

  int tid = threadIdx.x;
  { // stage all packed weights (86016 B) into LDS
    uint4* s4 = (uint4*)smem;
    for (int i = tid; i < 5376; i += 256) s4[i] = wB[i];
  }
  __syncthreads();

  int n = blockIdx.x;
  int wave = tid >> 5, lane = tid & 31, half = lane >> 4, lr = lane & 15;
  __bf16* hb = hball + wave * 2048;

  float colmax[16];
  #pragma unroll
  for (int i = 0; i < 16; i++) colmax[i] = -3.0e38f;

  for (int t = 0; t < 4; t++) {
    int tile = wave * 4 + t;
    int row  = tile * 16 + lr;

    // ---- layer 1: A = [16 x 6 (pad 32)] ----
    v16bf a1 = {};
    if (half == 0) {
      const float* p = pcls + ((size_t)n * PPTS + row) * 6;
      #pragma unroll
      for (int c = 0; c < 6; c++) a1[c] = (__bf16)p[c];   // K=0..5; all other K zero-padded
    }
    #pragma unroll
    for (int nt = 0; nt < 4; nt++) {
      v8f acc = {};
      v16bf bf = *(const v16bf*)(B1s + (nt * 32 + lane) * 16);
      acc = wmma_bf(a1, bf, acc);
      int col = nt * 16 + lr;
      float bb = b1[col], gg = g1[col], tt = bt1[col];
      #pragma unroll
      for (int v = 0; v < 8; v++) {
        float x = (acc[v] + bb) * gg + tt;
        x = fmaxf(x, 0.f);
        hb[(v + half * 8) * 64 + col] = (__bf16)x;        // D-layout -> row-major LDS
      }
    }

    // ---- layer 2: [16x64] @ [64x128] ----
    v16bf a2[2];
    #pragma unroll
    for (int kc = 0; kc < 2; kc++) {
      const __bf16* q = hb + lr * 64 + kc * 32 + half * 8;
      v8bf lo = *(const v8bf*)q, hi = *(const v8bf*)(q + 16);
      v16bf av;
      #pragma unroll
      for (int e = 0; e < 8; e++) { av[e] = lo[e]; av[e + 8] = hi[e]; }
      a2[kc] = av;
    }
    #pragma unroll 2
    for (int nt = 0; nt < 8; nt++) {
      v8f acc = {};
      #pragma unroll
      for (int kc = 0; kc < 2; kc++) {
        v16bf bf = *(const v16bf*)(B2s + ((kc * 8 + nt) * 32 + lane) * 16);
        acc = wmma_bf(a2[kc], bf, acc);
      }
      int col = nt * 16 + lr;
      float bb = b2[col], gg = g2[col], tt = bt2[col];
      #pragma unroll
      for (int v = 0; v < 8; v++) {
        float x = (acc[v] + bb) * gg + tt;
        x = fmaxf(x, 0.f);
        hb[(v + half * 8) * 128 + col] = (__bf16)x;
      }
    }

    // ---- layer 3: [16x128] @ [128x256], affine BN, no relu, max-pool ----
    v16bf a3[4];
    #pragma unroll
    for (int kc = 0; kc < 4; kc++) {
      const __bf16* q = hb + lr * 128 + kc * 32 + half * 8;
      v8bf lo = *(const v8bf*)q, hi = *(const v8bf*)(q + 16);
      v16bf av;
      #pragma unroll
      for (int e = 0; e < 8; e++) { av[e] = lo[e]; av[e + 8] = hi[e]; }
      a3[kc] = av;
    }
    #pragma unroll 2
    for (int nt = 0; nt < 16; nt++) {
      v8f acc = {};
      #pragma unroll
      for (int kc = 0; kc < 4; kc++) {
        v16bf bf = *(const v16bf*)(B3s + ((kc * 16 + nt) * 32 + lane) * 16);
        acc = wmma_bf(a3[kc], bf, acc);
      }
      int col = nt * 16 + lr;
      float bb = b3[col], gg = g3[col], tt = bt3[col];
      float m = -3.0e38f;
      #pragma unroll
      for (int v = 0; v < 8; v++) {
        float x = (acc[v] + bb) * gg + tt;
        m = fmaxf(m, x);
      }
      m = fmaxf(m, __shfl_xor(m, 16, 32));   // combine row halves
      colmax[nt] = fmaxf(colmax[nt], m);
    }
  }

  if (half == 0) {
    #pragma unroll
    for (int nt = 0; nt < 16; nt++) featb[wave * 256 + nt * 16 + lr] = colmax[nt];
  }
  __syncthreads();
  {
    float m = featb[tid];
    #pragma unroll
    for (int w = 1; w < 8; w++) m = fmaxf(m, featb[w * 256 + tid]);
    feat[(size_t)n * 256 + tid] = m;
  }
}

// ---------------- per-object mean of feat (deg==16 makes GCN agg an object mean) ----------------
__global__ void objmean_kernel(const float* __restrict__ feat, float* __restrict__ meanf)
{
  int o = blockIdx.x, c = threadIdx.x;
  float s = 0.f;
  #pragma unroll
  for (int k = 0; k < 16; k++) s += feat[((size_t)o * 16 + k) * 256 + c];
  meanf[o * 256 + c] = s * (1.f / 16.f);
}

// ---------------- one 256->256 WMMA layer for the collapsed object head ----------------
__device__ __forceinline__ void layer256_bf(const v16bf a[8], const __bf16* __restrict__ B,
                                            const float* __restrict__ bias, bool relu,
                                            __bf16* buf, int half, int lr, int lane)
{
  #pragma unroll 4
  for (int nt = 0; nt < 16; nt++) {
    v8f acc = {};
    #pragma unroll
    for (int kc = 0; kc < 8; kc++) {
      v16bf bf = *(const v16bf*)(B + ((kc * 16 + nt) * 32 + lane) * 16);
      acc = wmma_bf(a[kc], bf, acc);
    }
    int col = nt * 16 + lr;
    float bb = bias[col];
    #pragma unroll
    for (int v = 0; v < 8; v++) {
      float x = acc[v] + bb;
      if (relu) x = fmaxf(x, 0.f);
      buf[(v + half * 8) * 256 + col] = (__bf16)x;
    }
  }
}

__device__ __forceinline__ void load_a_from_bf(const __bf16* rowp, int half, v16bf a[8])
{
  #pragma unroll
  for (int kc = 0; kc < 8; kc++) {
    const __bf16* q = rowp + kc * 32 + half * 8;
    v8bf lo = *(const v8bf*)q, hi = *(const v8bf*)(q + 16);
    v16bf av;
    #pragma unroll
    for (int e = 0; e < 8; e++) { av[e] = lo[e]; av[e + 8] = hi[e]; }
    a[kc] = av;
  }
}

// emb -> gcn1(relu) -> gcn2 -> folded-Wc1(relu) -> Wc2(relu) -> Wc3 -> tanh, per object
__global__ __launch_bounds__(32) void head_kernel(
  const float* __restrict__ meanf,
  const __bf16* __restrict__ Be,  const float* __restrict__ be,
  const __bf16* __restrict__ Bg1, const float* __restrict__ bg1,
  const __bf16* __restrict__ Bg2, const float* __restrict__ bg2,
  const __bf16* __restrict__ Bc1, const float* __restrict__ bc1,
  const __bf16* __restrict__ Bc2, const float* __restrict__ bc2,
  const float* __restrict__ Wc3,  const float* __restrict__ bc3,
  float* __restrict__ cobj)
{
  __shared__ __bf16 buf[16 * 256];
  __shared__ float  fbuf[16 * 256];
  int lane = threadIdx.x & 31;
  int half = lane >> 4, lr = lane & 15;
  int m0 = blockIdx.x * 16;

  v16bf a[8];
  { // A-fragments straight from f32 per-object mean features
    const float* rp = meanf + (size_t)(m0 + lr) * 256;
    #pragma unroll
    for (int kc = 0; kc < 8; kc++) {
      v16bf av;
      #pragma unroll
      for (int e = 0; e < 8; e++) {
        av[e]     = (__bf16)rp[kc * 32 + half * 8 + e];
        av[e + 8] = (__bf16)rp[kc * 32 + 16 + half * 8 + e];
      }
      a[kc] = av;
    }
  }

  layer256_bf(a, Be,  be,  false, buf, half, lr, lane); load_a_from_bf(buf + lr * 256, half, a);
  layer256_bf(a, Bg1, bg1, true,  buf, half, lr, lane); load_a_from_bf(buf + lr * 256, half, a);
  layer256_bf(a, Bg2, bg2, false, buf, half, lr, lane); load_a_from_bf(buf + lr * 256, half, a);
  layer256_bf(a, Bc1, bc1, true,  buf, half, lr, lane); load_a_from_bf(buf + lr * 256, half, a);

  // c2 layer -> f32 staging for the final 256-dot
  #pragma unroll 4
  for (int nt = 0; nt < 16; nt++) {
    v8f acc = {};
    #pragma unroll
    for (int kc = 0; kc < 8; kc++) {
      v16bf bf = *(const v16bf*)(Bc2 + ((kc * 16 + nt) * 32 + lane) * 16);
      acc = wmma_bf(a[kc], bf, acc);
    }
    int col = nt * 16 + lr;
    float bb = bc2[col];
    #pragma unroll
    for (int v = 0; v < 8; v++)
      fbuf[(v + half * 8) * 256 + col] = fmaxf(acc[v] + bb, 0.f);
  }

  if (half == 0) {
    float s = bc3[0];
    for (int k = 0; k < 256; k++) s += fbuf[lr * 256 + k] * Wc3[k];
    cobj[m0 + lr] = tanhf(s);
  }
}

// ---------------- broadcast object score into [512,16,16], zero diagonal ----------------
__global__ void out_kernel(const float* __restrict__ cobj, float* __restrict__ out)
{
  int o = blockIdx.x, t = threadIdx.x;
  int i = t >> 4, j = t & 15;
  out[o * 256 + t] = (i == j) ? 0.f : cobj[o];
}

extern "C" void kernel_launch(void* const* d_in, const int* in_sizes, int n_in,
                              void* d_out, int out_size, void* d_ws, size_t ws_size,
                              hipStream_t stream)
{
  const float* pcls = (const float*)d_in[0];
  const float* W1  = (const float*)d_in[1];  const float* b1  = (const float*)d_in[2];
  const float* g1  = (const float*)d_in[3];  const float* bt1 = (const float*)d_in[4];
  const float* W2  = (const float*)d_in[5];  const float* b2  = (const float*)d_in[6];
  const float* g2  = (const float*)d_in[7];  const float* bt2 = (const float*)d_in[8];
  const float* W3  = (const float*)d_in[9];  const float* b3  = (const float*)d_in[10];
  const float* g3  = (const float*)d_in[11]; const float* bt3 = (const float*)d_in[12];
  const float* We  = (const float*)d_in[13]; const float* be  = (const float*)d_in[14];
  const float* Wg1 = (const float*)d_in[15]; const float* bg1 = (const float*)d_in[16];
  const float* Wg2 = (const float*)d_in[17]; const float* bg2 = (const float*)d_in[18];
  const float* Wc1 = (const float*)d_in[19]; const float* bc1 = (const float*)d_in[20];
  const float* Wc2 = (const float*)d_in[21]; const float* bc2 = (const float*)d_in[22];
  const float* Wc3 = (const float*)d_in[23]; const float* bc3 = (const float*)d_in[24];

  char* ws = (char*)d_ws;
  float*  feat  = (float*)(ws + 0);          // 8192*256 f32
  float*  meanf = (float*)(ws + 8388608);    // 512*256 f32
  float*  cobj  = (float*)(ws + 8912896);    // 512 f32
  __bf16* wB    = (__bf16*)(ws + 8916992);   // packed B1|B2|B3 (43008 bf16)
  __bf16* B1    = wB;
  __bf16* B2    = B1 + 2048;
  __bf16* B3    = B2 + 8192;
  __bf16* Be    = (__bf16*)(ws + 9003008);   // 5 x 65536 bf16
  __bf16* Bg1   = Be  + 65536;
  __bf16* Bg2   = Bg1 + 65536;
  __bf16* Bc1   = Bg2 + 65536;
  __bf16* Bc2   = Bc1 + 65536;

  pack_b_kernel<<<(2048 + 255) / 256, 256, 0, stream>>>(B1, W1, nullptr, 6, 4, 64, 2048);
  pack_b_kernel<<<(8192 + 255) / 256, 256, 0, stream>>>(B2, W2, nullptr, 64, 8, 128, 8192);
  pack_b_kernel<<<(32768 + 255) / 256, 256, 0, stream>>>(B3, W3, nullptr, 128, 16, 256, 32768);
  pack_b_kernel<<<256, 256, 0, stream>>>(Be,  We,  nullptr,           256, 16, 256, 65536);
  pack_b_kernel<<<256, 256, 0, stream>>>(Bg1, Wg1, nullptr,           256, 16, 256, 65536);
  pack_b_kernel<<<256, 256, 0, stream>>>(Bg2, Wg2, nullptr,           256, 16, 256, 65536);
  pack_b_kernel<<<256, 256, 0, stream>>>(Bc1, Wc1, Wc1 + 256 * 256,   256, 16, 256, 65536); // fold concat
  pack_b_kernel<<<256, 256, 0, stream>>>(Bc2, Wc2, nullptr,           256, 16, 256, 65536);

  size_t smem = (size_t)(2048 + 8192 + 32768 + 8 * 2048) * sizeof(__bf16) + 8 * 256 * sizeof(float); // 124 KB
  pointnet_kernel<<<8192, 256, smem, stream>>>(pcls, b1, g1, bt1, b2, g2, bt2, b3, g3, bt3,
                                               (const uint4*)wB, feat);
  objmean_kernel<<<512, 256, 0, stream>>>(feat, meanf);
  head_kernel<<<32, 32, 0, stream>>>(meanf, Be, be, Bg1, bg1, Bg2, bg2, Bc1, bc1, Bc2, bc2,
                                     Wc3, bc3, cobj);
  out_kernel<<<512, 256, 0, stream>>>(cobj, (float*)d_out);

  (void)in_sizes; (void)n_in; (void)out_size; (void)ws_size;
}
